// GroupedQueryAttention_45234595561823
// MI455X (gfx1250) — matrix operand order repaired and checked by reading, not compiled, not run on hardware
//
#include <hip/hip_runtime.h>
#include <cstdint>
#include <cstddef>

typedef __bf16 bf16_t;
typedef __attribute__((ext_vector_type(16))) __bf16 v16bf;
typedef __attribute__((ext_vector_type(8)))  float  v8f;

#define EMBED_DIM 2048
#define NUM_HEADS 32
#define NUM_KV_HEADS 8
#define HEAD_DIM 64
#define SEQ 2048
#define BATCH 2
#define KV_DIM (NUM_KV_HEADS * HEAD_DIM)

// Build a 16-element bf16 fragment from two 16-byte chunks.
static __device__ __forceinline__ v16bf ld_frag(const bf16_t* p0, const bf16_t* p1) {
  union U { uint4 q[2]; v16bf v; } u;
  u.q[0] = *(const uint4*)p0;
  u.q[1] = *(const uint4*)p1;
  return u.v;
}

// ---------- GEMM: C(MxN) = A(MxK) * W(KxN) + bias, bf16 WMMA, fp32 accum ----
// Block: 256 threads (8 waves), tile 64x256, K-step 32.
// Waves arranged 2(M) x 4(N); each wave owns 32x64 => 2x4 WMMA accumulators.
template <bool A_F32, bool OUT_F32>
__global__ __launch_bounds__(256) void gemm_bf16_wmma(
    const float* __restrict__ Af32,   // A in fp32 (converted to bf16) ...
    const bf16_t* __restrict__ Abf,   // ... or A already bf16
    const float* __restrict__ W,      // K x N, fp32 (converted to bf16)
    const float* __restrict__ bias,   // N
    float* __restrict__ Cf32,         // output fp32 ...
    bf16_t* __restrict__ Cbf,         // ... or output bf16
    const int M, const int N, const int K)
{
  __shared__ __align__(16) bf16_t sA[64][40];    // 64 x 32, padded rows (80B)
  __shared__ __align__(16) bf16_t sB[256][40];   // W tile transposed: [n][k]

  const int tid   = threadIdx.x;
  const int wid   = tid >> 5;
  const int lane  = tid & 31;
  const int lrow  = lane & 15;
  const int halfw = lane >> 4;
  const int wm = (wid >> 2) * 32;   // 0 or 32
  const int wn = (wid & 3) * 64;    // 0,64,128,192
  const int bm = blockIdx.x * 64;
  const int bn = blockIdx.y * 256;

  v8f acc[2][4] = {};

  for (int k0 = 0; k0 < K; k0 += 32) {
    { // stage A tile 64x32 (16B per thread)
      const int r = tid >> 2;
      const int c = (tid & 3) * 8;
      if constexpr (A_F32) {
        const float* src = Af32 + (size_t)(bm + r) * K + k0 + c;
        if (k0 + 32 < K) __builtin_prefetch(src + 32, 0, 1);
        #pragma unroll
        for (int i = 0; i < 8; ++i) sA[r][c + i] = (bf16_t)src[i];
      } else {
        // CDNA5 async copy: global -> LDS without VGPR round trip (ASYNCcnt).
        const bf16_t* src = Abf + (size_t)(bm + r) * K + k0 + c;
        unsigned ldsoff = (unsigned)(uintptr_t)&sA[r][c];
        asm volatile("global_load_async_to_lds_b128 %0, %1, off"
                     :: "v"(ldsoff), "v"(src) : "memory");
      }
    }
    { // stage W tile 32x256, transposed into sB[n][k] (32 elems/thread)
      const int kk = tid >> 3;           // 0..31
      const int c  = (tid & 7) * 32;     // 0..224
      const float* src = W + (size_t)(k0 + kk) * N + bn + c;
      if (k0 + 32 < K) __builtin_prefetch(src + (size_t)32 * N, 0, 1);
      #pragma unroll
      for (int i = 0; i < 32; ++i) sB[c + i][kk] = (bf16_t)src[i];
    }
    if constexpr (!A_F32) {
      asm volatile("s_wait_asynccnt 0x0" ::: "memory");
    }
    __syncthreads();

    v16bf afrag[2], bfrag[4];
    // A frag (16x32): lanes 0-15 rows M, K {0..7,16..23}; lanes 16-31 K {8..15,24..31}
    #pragma unroll
    for (int mi = 0; mi < 2; ++mi) {
      const bf16_t* ap = &sA[wm + mi * 16 + lrow][halfw * 8];
      afrag[mi] = ld_frag(ap, ap + 16);
    }
    // B frag (32x16): lanes 0-15 cols N, K {0..7,16..23}; lanes 16-31 K {8..15,24..31}
    #pragma unroll
    for (int ni = 0; ni < 4; ++ni) {
      const bf16_t* bp = &sB[wn + ni * 16 + lrow][halfw * 8];
      bfrag[ni] = ld_frag(bp, bp + 16);
    }
    #pragma unroll
    for (int mi = 0; mi < 2; ++mi)
      #pragma unroll
      for (int ni = 0; ni < 4; ++ni)
        acc[mi][ni] = __builtin_amdgcn_wmma_f32_16x16x32_bf16(
            false, afrag[mi], false, bfrag[ni], (short)0, acc[mi][ni], false, false);
    __syncthreads();
  }

  // Epilogue: C layout — VGPR j: M = j + 8*halfw, N = lrow.
  #pragma unroll
  for (int mi = 0; mi < 2; ++mi)
    #pragma unroll
    for (int ni = 0; ni < 4; ++ni) {
      const int n = bn + wn + ni * 16 + lrow;
      const float bv = bias[n];
      #pragma unroll
      for (int j = 0; j < 8; ++j) {
        const int m = bm + wm + mi * 16 + j + halfw * 8;
        const float v = acc[mi][ni][j] + bv;
        if constexpr (OUT_F32) Cf32[(size_t)m * N + n] = v;
        else                   Cbf[(size_t)m * N + n] = (bf16_t)v;
      }
    }
}

// ---------- RoPE (in place on bf16 Q or K in workspace) ----------
__global__ __launch_bounds__(256) void rope_kernel(bf16_t* __restrict__ t, const int nheads)
{
  const size_t total = (size_t)BATCH * SEQ * nheads * (HEAD_DIM / 2);
  size_t idx = (size_t)blockIdx.x * blockDim.x + threadIdx.x;
  if (idx >= total) return;
  const int i = (int)(idx & 31);          // freq index (half = 32)
  size_t idx2 = idx >> 5;
  const int h = (int)(idx2 % nheads);
  size_t idx3 = idx2 / nheads;
  const int s = (int)(idx3 % SEQ);
  const int b = (int)(idx3 / SEQ);
  bf16_t* p = t + (((size_t)b * SEQ + s) * nheads + h) * HEAD_DIM;
  const float t1 = (float)p[i];
  const float t2 = (float)p[i + 32];
  const float invf = __powf(10000.0f, -(float)i / 32.0f);
  const float ang = (float)s * invf;
  const float c = __cosf(ang), sn = __sinf(ang);
  p[i]      = (bf16_t)(t1 * c - t2 * sn);
  p[i + 32] = (bf16_t)(t1 * sn + t2 * c);
}

// ---------- Flash attention (causal, GQA), bf16 WMMA, fp32 softmax ----------
// Grid: (SEQ/64, BATCH*NUM_HEADS). Block: 128 threads (4 waves).
// Wave owns 16 query rows; KV processed in chunks of 32.
__global__ __launch_bounds__(128) void attn_kernel(
    const bf16_t* __restrict__ Q,   // (B, S, 2048)
    const bf16_t* __restrict__ Kc,  // (B, S, 512)
    const bf16_t* __restrict__ Vc,  // (B, S, 512)
    bf16_t* __restrict__ O)         // (B, S, 2048)
{
  __shared__ __align__(16) bf16_t sP[4][16][40];   // per-wave P tile 16x32
  __shared__ __align__(16) bf16_t sVT[64][40];     // V chunk transposed [d][k]

  const int qt  = blockIdx.x;
  const int b   = blockIdx.y >> 5;
  const int h   = blockIdx.y & 31;
  const int kvh = h >> 2;                 // 4 query heads per KV head
  const int q0  = qt * 64;

  const int tid   = threadIdx.x;
  const int wid   = tid >> 5;
  const int lane  = tid & 31;
  const int lrow  = lane & 15;
  const int halfw = lane >> 4;
  const int qrow0 = q0 + wid * 16;

  const bf16_t* Qb = Q  + ((size_t)b * SEQ) * EMBED_DIM + (size_t)h   * HEAD_DIM;
  const bf16_t* Kb = Kc + ((size_t)b * SEQ) * KV_DIM    + (size_t)kvh * HEAD_DIM;
  const bf16_t* Vb = Vc + ((size_t)b * SEQ) * KV_DIM    + (size_t)kvh * HEAD_DIM;

  // Q fragments, resident in registers for the whole tile (Dh=64 -> 2 K-steps)
  v16bf qfrag[2];
  {
    const bf16_t* qr = Qb + (size_t)(qrow0 + lrow) * EMBED_DIM + halfw * 8;
    qfrag[0] = ld_frag(qr,      qr + 16);
    qfrag[1] = ld_frag(qr + 32, qr + 48);
  }

  v8f acc[4] = {};
  float m_i[8], l_i[8];
  #pragma unroll
  for (int j = 0; j < 8; ++j) { m_i[j] = -3.0e38f; l_i[j] = 0.0f; }
  const float scale = 0.125f;   // 1/sqrt(64)

  for (int kv0 = 0; kv0 < q0 + 64; kv0 += 32) {
    // scores: Q(16x64) @ K^T(64x32). B-column n = key row of K (contiguous).
    v8f sc[2] = {};
    #pragma unroll
    for (int nt = 0; nt < 2; ++nt) {
      const bf16_t* kr = Kb + (size_t)(kv0 + nt * 16 + lrow) * KV_DIM + halfw * 8;
      v16bf kf0 = ld_frag(kr,      kr + 16);   // head-dims 0..31
      v16bf kf1 = ld_frag(kr + 32, kr + 48);   // head-dims 32..63
      sc[nt] = __builtin_amdgcn_wmma_f32_16x16x32_bf16(
          false, qfrag[0], false, kf0, (short)0, sc[nt], false, false);
      sc[nt] = __builtin_amdgcn_wmma_f32_16x16x32_bf16(
          false, qfrag[1], false, kf1, (short)0, sc[nt], false, false);
    }

    __syncthreads();  // prior PV reads of sVT done in all waves
    { // stage V chunk (32 keys x 64 dims) transposed into LDS (16 elems/thread)
      const int k  = tid >> 2;
      const int d0 = (tid & 3) * 16;
      const bf16_t* vr = Vb + (size_t)(kv0 + k) * KV_DIM + d0;
      #pragma unroll
      for (int i = 0; i < 16; ++i) sVT[d0 + i][k] = vr[i];
    }

    // Online softmax update. Row M = j + 8*halfw, key N = kv0 + nt*16 + lrow.
    float pf[2][8], alpha[8];
    #pragma unroll
    for (int j = 0; j < 8; ++j) {
      const int row = qrow0 + j + halfw * 8;
      float s0 = sc[0][j] * scale;
      float s1 = sc[1][j] * scale;
      if (kv0 + lrow      > row) s0 = -3.0e38f;   // causal mask
      if (kv0 + 16 + lrow > row) s1 = -3.0e38f;
      float rm = fmaxf(s0, s1);
      #pragma unroll
      for (int msk = 1; msk < 16; msk <<= 1) rm = fmaxf(rm, __shfl_xor(rm, msk, 32));
      const float mnew = fmaxf(m_i[j], rm);
      alpha[j] = __expf(m_i[j] - mnew);
      m_i[j] = mnew;
      const float p0 = __expf(s0 - mnew);
      const float p1 = __expf(s1 - mnew);
      pf[0][j] = p0; pf[1][j] = p1;
      float rs = p0 + p1;
      #pragma unroll
      for (int msk = 1; msk < 16; msk <<= 1) rs += __shfl_xor(rs, msk, 32);
      l_i[j] = l_i[j] * alpha[j] + rs;
    }
    #pragma unroll
    for (int nt = 0; nt < 4; ++nt)
      #pragma unroll
      for (int j = 0; j < 8; ++j)
        acc[nt][j] *= alpha[j];

    // Reformat P from C-layout to A-layout via this wave's LDS region.
    #pragma unroll
    for (int nt = 0; nt < 2; ++nt)
      #pragma unroll
      for (int j = 0; j < 8; ++j)
        sP[wid][j + halfw * 8][nt * 16 + lrow] = (bf16_t)pf[nt][j];
    __syncthreads();  // sVT writes + sP writes visible

    // acc += P(16x32) @ V(32x64)
    {
      const bf16_t* pr = &sP[wid][lrow][halfw * 8];
      v16bf pfrag = ld_frag(pr, pr + 16);
      #pragma unroll
      for (int nt = 0; nt < 4; ++nt) {
        const bf16_t* vp = &sVT[nt * 16 + lrow][halfw * 8];
        v16bf vf = ld_frag(vp, vp + 16);
        acc[nt] = __builtin_amdgcn_wmma_f32_16x16x32_bf16(
            false, pfrag, false, vf, (short)0, acc[nt], false, false);
      }
    }
  }

  // Normalize and write (B, S, 2048) bf16
  float inv[8];
  #pragma unroll
  for (int j = 0; j < 8; ++j) inv[j] = 1.0f / l_i[j];
  #pragma unroll
  for (int nt = 0; nt < 4; ++nt)
    #pragma unroll
    for (int j = 0; j < 8; ++j) {
      const int row = qrow0 + j + halfw * 8;
      const int d   = nt * 16 + lrow;
      O[((size_t)b * SEQ + row) * EMBED_DIM + (size_t)h * HEAD_DIM + d] =
          (bf16_t)(acc[nt][j] * inv[j]);
    }
}

// ---------- host-side launch ----------
extern "C" void kernel_launch(void* const* d_in, const int* in_sizes, int n_in,
                              void* d_out, int out_size, void* d_ws, size_t ws_size,
                              hipStream_t stream) {
  (void)in_sizes; (void)n_in; (void)out_size; (void)ws_size;
  const float* x  = (const float*)d_in[0];
  const float* wq = (const float*)d_in[1];
  const float* bq = (const float*)d_in[2];
  const float* wk = (const float*)d_in[3];
  const float* bk = (const float*)d_in[4];
  const float* wv = (const float*)d_in[5];
  const float* bv = (const float*)d_in[6];
  const float* wo = (const float*)d_in[7];
  const float* bo = (const float*)d_in[8];
  float* out = (float*)d_out;

  char* ws = (char*)d_ws;
  bf16_t* qbuf = (bf16_t*)(ws);                                // 16 MiB
  bf16_t* kbuf = (bf16_t*)(ws + (size_t)16 * 1024 * 1024);     //  4 MiB
  bf16_t* vbuf = (bf16_t*)(ws + (size_t)20 * 1024 * 1024);     //  4 MiB
  bf16_t* abuf = (bf16_t*)(ws + (size_t)24 * 1024 * 1024);     // 16 MiB

  const int M = BATCH * SEQ;   // 4096
  dim3 blk(256);

  gemm_bf16_wmma<true, false><<<dim3(M / 64, EMBED_DIM / 256), blk, 0, stream>>>(
      x, nullptr, wq, bq, nullptr, qbuf, M, EMBED_DIM, EMBED_DIM);
  gemm_bf16_wmma<true, false><<<dim3(M / 64, KV_DIM / 256), blk, 0, stream>>>(
      x, nullptr, wk, bk, nullptr, kbuf, M, KV_DIM, EMBED_DIM);
  gemm_bf16_wmma<true, false><<<dim3(M / 64, KV_DIM / 256), blk, 0, stream>>>(
      x, nullptr, wv, bv, nullptr, vbuf, M, KV_DIM, EMBED_DIM);

  {
    size_t tq = (size_t)BATCH * SEQ * NUM_HEADS * (HEAD_DIM / 2);
    rope_kernel<<<dim3((unsigned)((tq + 255) / 256)), blk, 0, stream>>>(qbuf, NUM_HEADS);
    size_t tk = (size_t)BATCH * SEQ * NUM_KV_HEADS * (HEAD_DIM / 2);
    rope_kernel<<<dim3((unsigned)((tk + 255) / 256)), blk, 0, stream>>>(kbuf, NUM_KV_HEADS);
  }

  attn_kernel<<<dim3(SEQ / 64, BATCH * NUM_HEADS), dim3(128), 0, stream>>>(
      qbuf, kbuf, vbuf, abuf);

  gemm_bf16_wmma<false, true><<<dim3(M / 64, EMBED_DIM / 256), blk, 0, stream>>>(
      nullptr, abuf, wo, bo, out, nullptr, M, EMBED_DIM, EMBED_DIM);
}
